// TagGNN_QI_24756191494706
// MI455X (gfx1250) — matrix operand order repaired and checked by reading, not compiled, not run on hardware
//
#include <hip/hip_runtime.h>
#include <hip/hip_bf16.h>

// ---------------------------------------------------------------------------
// TagGNN pipeline for MI455X (gfx1250, wave32):
//   h0 = mean-pool(word_emb[wids]) / lengths
//   h1 = leaky_relu(h0 @ W1_self + mean_neigh(h0) @ W1_neigh + b1)
//   out = h1 @ W2_self + mean_neigh(h1) @ W2_neigh + b2
// GEMMs via v_wmma_f32_16x16x32_bf16 (bf16 in / f32 accumulate).
// ---------------------------------------------------------------------------

#define NN   102000
#define DD   128
#define EE   1000000
#define LL   12
#define NTILES (NN / 16)     // 6375, exact
#define NEG_SLOPE 0.01f

typedef __attribute__((ext_vector_type(16))) __bf16 v16bf;
typedef __attribute__((ext_vector_type(8)))  __bf16 v8bf;
typedef __attribute__((ext_vector_type(8)))  float  v8f;

// ---------------------------------------------------------------------------
__global__ void k_zero(float* __restrict__ p, long n) {
    long stride = (long)gridDim.x * blockDim.x;
    for (long i = (long)blockIdx.x * blockDim.x + threadIdx.x; i < n; i += stride)
        p[i] = 0.0f;
}

// ---------------------------------------------------------------------------
// h0[n][d] = sum_l word_emb[wids[n][l]][d] / lengths[n]
__global__ void k_embed(const int* __restrict__ wids,
                        const float* __restrict__ lengths,
                        const float* __restrict__ we,
                        float* __restrict__ h0) {
    int d = threadIdx.x;  // 128 threads = 128 dims
    for (int n = blockIdx.x; n < NN; n += gridDim.x) {
        float inv = 1.0f / lengths[n];
        float acc = 0.0f;
#pragma unroll
        for (int l = 0; l < LL; ++l) {
            int w = wids[n * LL + l];
            acc += we[(long)w * DD + d];
        }
        h0[(long)n * DD + d] = acc * inv;
    }
}

// ---------------------------------------------------------------------------
__global__ void k_degree(const int* __restrict__ dst, float* __restrict__ deg) {
    int stride = gridDim.x * blockDim.x;
    for (int e = blockIdx.x * blockDim.x + threadIdx.x; e < EE; e += stride)
        __hip_atomic_fetch_add(&deg[dst[e]], 1.0f,
                               __ATOMIC_RELAXED, __HIP_MEMORY_SCOPE_AGENT);
}

// ---------------------------------------------------------------------------
// hn[dst] += h[src] over all edges.  256 threads = 2 edges x 128 dims.
__global__ void k_scatter(const int* __restrict__ src,
                          const int* __restrict__ dst,
                          const float* __restrict__ h,
                          float* __restrict__ hn) {
    int d    = threadIdx.x & 127;
    int slot = threadIdx.x >> 7;                 // 0 or 1
    long estep = (long)gridDim.x * 2;
    for (long e = (long)blockIdx.x * 2 + slot; e < EE; e += estep) {
        int s = src[e];
        int t = dst[e];
        // gfx1250 prefetch of next source row (global_prefetch_b8)
        long en = e + estep;
        if (en < EE)
            __builtin_prefetch(&h[(long)src[en] * DD], 0, 0);
        float v = h[(long)s * DD + d];
        __hip_atomic_fetch_add(&hn[(long)t * DD + d], v,
                               __ATOMIC_RELAXED, __HIP_MEMORY_SCOPE_AGENT);
    }
}

// ---------------------------------------------------------------------------
// B fragment (32x16 bf16, column-major per lane) from LDS weight stored as
// W^T: ldsW[n*128 + k].  lane%16 = column N; lane half selects K 0-15 / 16-31.
__device__ __forceinline__ v16bf load_b_frag(const __bf16* __restrict__ ldsW,
                                             int colBase, int kBase, int lane) {
    int col = colBase + (lane & 15);
    int k   = kBase + ((lane & 16) ? 16 : 0);
    const v8bf* p = (const v8bf*)(ldsW + col * DD + k);   // 32B-aligned
    v8bf lo = p[0];
    v8bf hi = p[1];
    return __builtin_shufflevector(lo, hi,
        0, 1, 2, 3, 4, 5, 6, 7, 8, 9, 10, 11, 12, 13, 14, 15);
}

// A fragment (16x32 bf16) built from one f32 row, scaled.
// Layout: lane<16 -> K = kBase+{0..15}; lane>=16 -> K = kBase+{16..31}.
__device__ __forceinline__ v16bf make_a_frag(const float* __restrict__ rowPtr,
                                             int kBase, int lane, float scale) {
    int off = (lane & 16) ? 16 : 0;
    const float4* p0 = (const float4*)(rowPtr + kBase + off);
    const float4* p1 = (const float4*)(rowPtr + kBase + off + 8);
    float4 a = p0[0], b = p0[1];
    float4 c = p1[0], d = p1[1];
    v16bf r;
    r[0]  = (__bf16)(a.x * scale); r[1]  = (__bf16)(a.y * scale);
    r[2]  = (__bf16)(a.z * scale); r[3]  = (__bf16)(a.w * scale);
    r[4]  = (__bf16)(b.x * scale); r[5]  = (__bf16)(b.y * scale);
    r[6]  = (__bf16)(b.z * scale); r[7]  = (__bf16)(b.w * scale);
    r[8]  = (__bf16)(c.x * scale); r[9]  = (__bf16)(c.y * scale);
    r[10] = (__bf16)(c.z * scale); r[11] = (__bf16)(c.w * scale);
    r[12] = (__bf16)(d.x * scale); r[13] = (__bf16)(d.y * scale);
    r[14] = (__bf16)(d.z * scale); r[15] = (__bf16)(d.w * scale);
    return r;
}

// ---------------------------------------------------------------------------
// out = act( h @ Wself + (hn/max(deg,1)) @ Wneigh + bias )
// 128 threads = 4 waves; each wave owns one 16-row tile per iteration.
// launch_bounds(128, 4): target >=4 waves/SIMD (<=256 VGPRs/wave).
__global__ __launch_bounds__(128, 4)
void k_sage_wmma(const float* __restrict__ h,
                 const float* __restrict__ hn,
                 const float* __restrict__ deg,
                 const float* __restrict__ Wself,
                 const float* __restrict__ Wneigh,
                 const float* __restrict__ bias,
                 float* __restrict__ out,
                 int applyAct) {
    __shared__ __bf16 sWs[DD * DD];   // W_self^T  in bf16 (32 KB)
    __shared__ __bf16 sWn[DD * DD];   // W_neigh^T in bf16 (32 KB)

    // Stage weights transposed: sW[n*128 + k] = W[k*128 + n]
    for (int i = threadIdx.x; i < DD * DD; i += blockDim.x) {
        int k = i >> 7;
        int n = i & 127;
        sWs[n * DD + k] = (__bf16)Wself[i];
        sWn[n * DD + k] = (__bf16)Wneigh[i];
    }
    __syncthreads();

    int lane = threadIdx.x & 31;
    int wave = threadIdx.x >> 5;

    for (int tile = blockIdx.x * 4 + wave; tile < NTILES; tile += gridDim.x * 4) {
        int arow = tile * 16 + (lane & 15);          // row this lane feeds into A
        const float* hrow  = h  + (long)arow * DD;
        const float* hnrow = hn + (long)arow * DD;
        // deg holds an integer count in f32: fast v_rcp_f32 is plenty here
        // and avoids the full IEEE divide expansion inside the tile loop.
        float nscale = __builtin_amdgcn_rcpf(fmaxf(deg[arow], 1.0f));

        v8f acc[8] = {};

        // unroll-by-2: two K-slices in flight gives ILP without blowing the
        // register file (keeps ~4 waves/SIMD instead of ~1 at full unroll).
#pragma unroll 2
        for (int ki = 0; ki < 4; ++ki) {
            int kBase = ki * 32;
            v16bf aS = make_a_frag(hrow,  kBase, lane, 1.0f);
            v16bf aN = make_a_frag(hnrow, kBase, lane, nscale);
#pragma unroll
            for (int ct = 0; ct < 8; ++ct) {
                v16bf bS = load_b_frag(sWs, ct * 16, kBase, lane);
                acc[ct] = __builtin_amdgcn_wmma_f32_16x16x32_bf16(
                    false, aS, false, bS, (short)0, acc[ct], false, false);
                v16bf bN = load_b_frag(sWn, ct * 16, kBase, lane);
                acc[ct] = __builtin_amdgcn_wmma_f32_16x16x32_bf16(
                    false, aN, false, bN, (short)0, acc[ct], false, false);
            }
        }

        // Epilogue.  C/D layout: VGPR r -> row r (lanes 0-15) / row r+8
        // (lanes 16-31); column = ct*16 + lane%16.
        int col0 = lane & 15;
        int mofs = (lane & 16) ? 8 : 0;
#pragma unroll
        for (int ct = 0; ct < 8; ++ct) {
            int col = ct * 16 + col0;
            float bv = bias[col];
#pragma unroll
            for (int r = 0; r < 8; ++r) {
                float o = acc[ct][r] + bv;
                if (applyAct) o = (o > 0.0f) ? o : NEG_SLOPE * o;
                out[(long)(tile * 16 + mofs + r) * DD + col] = o;
            }
        }
    }
}

// ---------------------------------------------------------------------------
extern "C" void kernel_launch(void* const* d_in, const int* in_sizes, int n_in,
                              void* d_out, int out_size, void* d_ws, size_t ws_size,
                              hipStream_t stream) {
    const int*   wids    = (const int*)  d_in[0];
    const float* lengths = (const float*)d_in[1];
    const int*   src     = (const int*)  d_in[2];
    const int*   dst     = (const int*)  d_in[3];
    const float* wemb    = (const float*)d_in[4];
    const float* W1s     = (const float*)d_in[5];
    const float* W1n     = (const float*)d_in[6];
    const float* b1      = (const float*)d_in[7];
    const float* W2s     = (const float*)d_in[8];
    const float* W2n     = (const float*)d_in[9];
    const float* b2      = (const float*)d_in[10];
    float* out = (float*)d_out;

    // Workspace layout (all f32): h0 | hn | h1 | deg
    float* h0  = (float*)d_ws;
    float* hnb = h0  + (long)NN * DD;
    float* h1  = hnb + (long)NN * DD;
    float* deg = h1  + (long)NN * DD;

    // 0) init accumulators
    k_zero<<<2048, 256, 0, stream>>>(hnb, (long)NN * DD);
    k_zero<<<256, 256, 0, stream>>>(deg, (long)NN);

    // 1) embedding mean-pool
    k_embed<<<8192, 128, 0, stream>>>(wids, lengths, wemb, h0);

    // 2) in-degree (reused for both layers)
    k_degree<<<2048, 256, 0, stream>>>(dst, deg);

    // 3) layer 1 neighbor sum + fused SAGE GEMM (leaky_relu)
    k_scatter<<<8192, 256, 0, stream>>>(src, dst, h0, hnb);
    k_sage_wmma<<<1024, 128, 0, stream>>>(h0, hnb, deg, W1s, W1n, b1, h1, 1);

    // 4) layer 2 neighbor sum + fused SAGE GEMM (no activation)
    k_zero<<<2048, 256, 0, stream>>>(hnb, (long)NN * DD);
    k_scatter<<<8192, 256, 0, stream>>>(src, dst, h1, hnb);
    k_sage_wmma<<<1024, 128, 0, stream>>>(h1, hnb, deg, W2s, W2n, b2, out, 0);
}
